// SelfAttention_19894288515569
// MI455X (gfx1250) — matrix operand order repaired and checked by reading, not compiled, not run on hardware
//
#include <hip/hip_runtime.h>
#include <hip/hip_bf16.h>

typedef __attribute__((ext_vector_type(16))) _Float16 v16h;
typedef __attribute__((ext_vector_type(4)))  _Float16 v4h;
typedef __attribute__((ext_vector_type(8)))  float    v8f;
typedef __attribute__((ext_vector_type(4)))  int      v4i;

#define HIDDEN 1024
#define HEADS  16
#define HDIM   64
#define BATCH  2
#define SEQ    2048

// ---------------- CDNA5 async global->LDS copy (ASYNCcnt-tracked) ----------
#if defined(__has_builtin)
#if __has_builtin(__builtin_amdgcn_global_load_async_to_lds_b128)
#define HAVE_ASYNC_COPY 1
#endif
#endif
#ifndef HAVE_ASYNC_COPY
#define HAVE_ASYNC_COPY 0
#endif

typedef __attribute__((address_space(1))) v4i g_v4i;  // global <4 x i32>
typedef __attribute__((address_space(3))) v4i l_v4i;  // LDS    <4 x i32>

// 16-byte (b128) chunk: global -> LDS, no VGPR data path.
__device__ __forceinline__ void async_b128(void* lds_dst, const void* gsrc) {
#if HAVE_ASYNC_COPY
  __builtin_amdgcn_global_load_async_to_lds_b128(
      (g_v4i*)(uintptr_t)gsrc,
      (l_v4i*)(uint32_t)(uintptr_t)lds_dst, 0, 0);
#else
  *(uint4*)lds_dst = *(const uint4*)gsrc;  // synchronous fallback
#endif
}

template <int N>
__device__ __forceinline__ void wait_async() {
#if HAVE_ASYNC_COPY
#if __has_builtin(__builtin_amdgcn_s_wait_asynccnt)
  __builtin_amdgcn_s_wait_asynccnt(N);
#else
  asm volatile("s_wait_asynccnt %0" ::"i"(N) : "memory");
#endif
#endif
}

// ---------------- WMMA helpers (CDNA5 wave32 VGPR layouts) ------------------
__device__ __forceinline__ v8f wmma_f16(v16h a, v16h b, v8f c) {
  return __builtin_amdgcn_wmma_f32_16x16x32_f16(false, a, false, b, (short)0, c,
                                                false, false);
}

// A tile 16x32 (MxK) from row-major f16, leading dim ld.
__device__ __forceinline__ v16h load_A(const _Float16* src, int ld, int lane) {
  const int row = lane & 15;
  const int kb  = (lane >> 4) << 4;
  const _Float16* p = src + (size_t)row * ld + kb;
  v16h a;
#pragma unroll
  for (int i = 0; i < 8; ++i) a[i] = p[i];
#pragma unroll
  for (int i = 0; i < 8; ++i) a[8 + i] = p[8 + i];
  return a;
}

// Same, converting from f32 source.
__device__ __forceinline__ v16h load_A_f32(const float* src, int ld, int lane) {
  const int row = lane & 15;
  const int kb  = (lane >> 4) << 4;
  const float* p = src + (size_t)row * ld + kb;
  v16h a;
#pragma unroll
  for (int i = 0; i < 8; ++i) a[i] = (_Float16)p[i];
#pragma unroll
  for (int i = 0; i < 8; ++i) a[8 + i] = (_Float16)p[8 + i];
  return a;
}

// B tile 32x16 (KxN) from "N-major" source: src[n*ld + k]. Contiguous per lane.
__device__ __forceinline__ v16h load_B_nk(const _Float16* src, int ld, int lane) {
  const int col = lane & 15;
  const int kb  = (lane >> 4) << 4;
  const _Float16* p = src + (size_t)col * ld + kb;
  v16h b;
#pragma unroll
  for (int i = 0; i < 16; ++i) b[i] = p[i];
  return b;
}

// B tile 32x16 from row-major KxN source (strided per lane).
__device__ __forceinline__ v16h load_B_kn(const _Float16* src, int ld, int lane) {
  const int col = lane & 15;
  const int kb  = (lane >> 4) << 4;
  v16h b;
#pragma unroll
  for (int i = 0; i < 16; ++i) b[i] = src[(size_t)(kb + i) * ld + col];
  return b;
}

// ---------------------------------------------------------------------
// Kernel 1: per-head projection  y[b,h,s,:] = x[b,s,h,:] @ W^T + bias
// ---------------------------------------------------------------------
__global__ __launch_bounds__(256) void head_proj_kernel(
    const float* __restrict__ x, const float* __restrict__ W,
    const float* __restrict__ bias, _Float16* __restrict__ y) {
  __shared__ _Float16 sW[HDIM * HDIM];
  __shared__ float    sB[HDIM];
  const int tid  = threadIdx.x;
  const int lane = tid & 31;
  const int wave = tid >> 5;

  for (int i = tid; i < HDIM * HDIM; i += 256) sW[i] = (_Float16)W[i];
  if (tid < HDIM) sB[tid] = bias[tid];
  __syncthreads();

  const int bid   = blockIdx.x;
  const int b     = bid >> 8;
  const int h     = (bid >> 4) & 15;
  const int chunk = bid & 15;
  const int s0    = chunk * 128 + wave * 16;

  const float* xrow = x + (size_t)(b * SEQ + s0) * HIDDEN + h * HDIM;
  v8f acc[4] = {};
#pragma unroll
  for (int kk = 0; kk < 2; ++kk) {
    v16h a = load_A_f32(xrow + kk * 32, HIDDEN, lane);
#pragma unroll
    for (int n = 0; n < 4; ++n) {
      v16h bm = load_B_nk(sW + (n * 16) * HDIM + kk * 32, HDIM, lane);
      acc[n] = wmma_f16(a, bm, acc[n]);
    }
  }
  const int col = lane & 15;
  _Float16* yb = y + (size_t)((b * HEADS + h) * SEQ + s0) * HDIM;
#pragma unroll
  for (int n = 0; n < 4; ++n) {
#pragma unroll
    for (int r = 0; r < 8; ++r) {
      const int row = (lane < 16) ? r : (r + 8);
      yb[(size_t)row * HDIM + n * 16 + col] =
          (_Float16)(acc[n][r] + sB[n * 16 + col]);
    }
  }
}

// ---------------------------------------------------------------------
// Kernel 2: flash attention. Block = (b, h, 128 q rows), 8 waves x 16 rows.
// K/V tiles triple-buffered via async global->LDS; 1 barrier per tile.
// ---------------------------------------------------------------------
__device__ __forceinline__ void stage_kv(_Float16* dK, _Float16* dV,
                                         const _Float16* gK,
                                         const _Float16* gV, int tid) {
  // 64x64 f16 tile = 512 b128 chunks; 2 K + 2 V chunks per thread.
#pragma unroll
  for (int i = 0; i < 2; ++i) {
    const int off = (tid + i * 256) * 8;
    async_b128(dK + off, gK + off);
    async_b128(dV + off, gV + off);
  }
}

__global__ __launch_bounds__(256) void flash_attn_kernel(
    const _Float16* __restrict__ qf, const _Float16* __restrict__ kf,
    const _Float16* __restrict__ vf, const int* __restrict__ mask,
    _Float16* __restrict__ out) {
  __shared__ _Float16 sK[3][64 * HDIM];   // 24KB
  __shared__ _Float16 sV[3][64 * HDIM];   // 24KB
  __shared__ _Float16 sP[8][16 * HDIM];   // per-wave transpose bounce, 16KB

  const int tid  = threadIdx.x;
  const int lane = tid & 31;
  const int wave = tid >> 5;
  const int col  = lane & 15;

  const int bid   = blockIdx.x;
  const int b     = bid >> 8;
  const int h     = (bid >> 4) & 15;
  const int chunk = bid & 15;
  const int qw    = chunk * 128 + wave * 16;

  const _Float16* qbase  = qf + (size_t)((b * HEADS + h) * SEQ + qw) * HDIM;
  const _Float16* kbase0 = kf + (size_t)((b * HEADS + h) * SEQ) * HDIM;
  const _Float16* vbase0 = vf + (size_t)((b * HEADS + h) * SEQ) * HDIM;
  const int* mrow = mask + (size_t)b * SEQ * SEQ;

  const v16h aq0 = load_A(qbase, HDIM, lane);
  const v16h aq1 = load_A(qbase + 32, HDIM, lane);

  v8f o[4] = {};
  float mx[8], ls[8];
#pragma unroll
  for (int r = 0; r < 8; ++r) { mx[r] = -3.4e38f; ls[r] = 0.f; }

  _Float16* pbuf = sP[wave];

  constexpr int NT = SEQ / 64;
  stage_kv(sK[0], sV[0], kbase0, vbase0, tid);

  for (int kt = 0; kt < NT; ++kt) {
    const int k0 = kt * 64;
    if (kt + 1 < NT) {
      const size_t goff = (size_t)(k0 + 64) * HDIM;
      stage_kv(sK[(kt + 1) % 3], sV[(kt + 1) % 3], kbase0 + goff,
               vbase0 + goff, tid);
      wait_async<4>();  // this wave's chunks of tile kt have landed
    } else {
      wait_async<0>();
    }
    __syncthreads();    // whole tile kt visible; also fences buffer reuse
    const _Float16* cK = sK[kt % 3];
    const _Float16* cV = sV[kt % 3];

    // hint: pull next mask tile toward L2/L1 (one line per lane)
    if (kt + 1 < NT)
      __builtin_prefetch(mrow + (size_t)(qw + (lane & 15)) * SEQ + k0 + 64, 0, 1);

    // S = Q K^T : B[k=d][n=key j] = K[j][d] -> N-major, contiguous per lane
    v8f sc[4] = {};
#pragma unroll
    for (int n = 0; n < 4; ++n) {
      v16h b0 = load_B_nk(cK + (n * 16) * HDIM, HDIM, lane);
      v16h b1 = load_B_nk(cK + (n * 16) * HDIM + 32, HDIM, lane);
      sc[n] = wmma_f16(aq0, b0, sc[n]);
      sc[n] = wmma_f16(aq1, b1, sc[n]);
    }

    // mask, scale (1/32), online softmax; rows live in 16-lane halves
#pragma unroll
    for (int r = 0; r < 8; ++r) {
      const int row = (lane < 16) ? r : (r + 8);
      const int qg  = qw + row;
      float tmax = -3.4e38f;
#pragma unroll
      for (int n = 0; n < 4; ++n) {
        const int kg = k0 + n * 16 + col;
        const int mv = mrow[(size_t)qg * SEQ + kg];
        const float t = (mv != 0) ? (sc[n][r] * 0.03125f) : -3.125e18f;
        sc[n][r] = t;
        tmax = fmaxf(tmax, t);
      }
      tmax = fmaxf(tmax, __shfl_xor(tmax, 1, 32));
      tmax = fmaxf(tmax, __shfl_xor(tmax, 2, 32));
      tmax = fmaxf(tmax, __shfl_xor(tmax, 4, 32));
      tmax = fmaxf(tmax, __shfl_xor(tmax, 8, 32));
      const float mnew = fmaxf(mx[r], tmax);
      const float corr = __expf(mx[r] - mnew);
      mx[r] = mnew;
      float psum = 0.f;
#pragma unroll
      for (int n = 0; n < 4; ++n) {
        const float p = __expf(sc[n][r] - mnew);
        psum += p;
        pbuf[(size_t)row * HDIM + n * 16 + col] = (_Float16)p;  // C-layout
        o[n][r] *= corr;
      }
      psum += __shfl_xor(psum, 1, 32);
      psum += __shfl_xor(psum, 2, 32);
      psum += __shfl_xor(psum, 4, 32);
      psum += __shfl_xor(psum, 8, 32);
      ls[r] = ls[r] * corr + psum;
    }
    // No barrier: sP region is private to this wave, and per-wave LDS ops
    // execute in order (DS pipe in-order per ISA) -> store->load is safe.

    // O += P @ V : A = P (16x64 row-major in LDS), B = V row-major KxN
#pragma unroll
    for (int kk = 0; kk < 2; ++kk) {
      v16h ap = load_A(pbuf + kk * 32, HDIM, lane);
#pragma unroll
      for (int n = 0; n < 4; ++n) {
        v16h bv = load_B_kn(cV + (size_t)(kk * 32) * HDIM + n * 16, HDIM, lane);
        o[n] = wmma_f16(ap, bv, o[n]);
      }
    }
    // No tail barrier: triple buffering gives reuse distance 3; the single
    // barrier at the top of each iteration already separates last reads of a
    // buffer from the async issue that overwrites it two iterations later.
  }

  _Float16* ob = out + (size_t)(b * SEQ + qw) * HIDDEN + h * HDIM;
#pragma unroll
  for (int r = 0; r < 8; ++r) {
    const int row = (lane < 16) ? r : (r + 8);
    const float rl = (ls[r] > 0.f) ? (1.f / ls[r]) : 0.f;
#pragma unroll
    for (int n = 0; n < 4; ++n) {
      ob[(size_t)row * HIDDEN + n * 16 + col] = (_Float16)(o[n][r] * rl);
    }
  }
}

// ---------------------------------------------------------------------
// Kernel 3a: Wo f32 -> f16 (layout preserved: [N=1024][K=1024], N-major)
// ---------------------------------------------------------------------
__global__ __launch_bounds__(256) void wcvt_kernel(const float* __restrict__ w,
                                                   _Float16* __restrict__ wf) {
  const size_t i = ((size_t)blockIdx.x * 256 + threadIdx.x) * 4;
  const float4 v = *(const float4*)(w + i);
  v4h o;
  o[0] = (_Float16)v.x; o[1] = (_Float16)v.y;
  o[2] = (_Float16)v.z; o[3] = (_Float16)v.w;
  *(v4h*)(wf + i) = o;
}

// ---------------------------------------------------------------------
// Kernel 3b: out = attn[4096,1024] @ Wo^T + bo (f32). Wo-f16 tiles are
// async-staged to LDS ([n][k], already N-major), triple buffered.
// ---------------------------------------------------------------------
__device__ __forceinline__ void stage_w(_Float16* buf, const _Float16* wsrc,
                                        int tid) {
  // 64n x 64k f16 tile = 512 b128 chunks; 2 per thread. Row = 8 chunks.
#pragma unroll
  for (int i = 0; i < 2; ++i) {
    const int c  = tid + i * 256;
    const int n  = c >> 3;
    const int ko = (c & 7) * 8;
    async_b128(buf + n * 64 + ko, wsrc + (size_t)n * HIDDEN + ko);
  }
}

__global__ __launch_bounds__(256) void out_proj_kernel(
    const _Float16* __restrict__ a, const _Float16* __restrict__ wf,
    const float* __restrict__ bo, float* __restrict__ out) {
  __shared__ _Float16 sB[3][64 * 64];  // [n][k] tiles, 24KB
  const int tid  = threadIdx.x;
  const int lane = tid & 31;
  const int wave = tid >> 5;
  const int col  = lane & 15;

  const int mblk = blockIdx.x >> 4;
  const int nblk = blockIdx.x & 15;
  const int m0   = mblk * 128 + wave * 16;
  const int n0   = nblk * 64;

  v8f acc[4] = {};
  constexpr int NT = HIDDEN / 64;
  const _Float16* wbase = wf + (size_t)n0 * HIDDEN;

  stage_w(sB[0], wbase, tid);
  for (int kt = 0; kt < NT; ++kt) {
    const int k0 = kt * 64;
    if (kt + 1 < NT) {
      stage_w(sB[(kt + 1) % 3], wbase + (k0 + 64), tid);
      wait_async<2>();
    } else {
      wait_async<0>();
    }
    __syncthreads();
    const _Float16* bt = sB[kt % 3];
#pragma unroll
    for (int kk = 0; kk < 2; ++kk) {
      v16h av = load_A(a + (size_t)m0 * HIDDEN + k0 + kk * 32, HIDDEN, lane);
#pragma unroll
      for (int n = 0; n < 4; ++n) {
        v16h bv = load_B_nk(bt + (n * 16) * 64 + kk * 32, 64, lane);
        acc[n] = wmma_f16(av, bv, acc[n]);
      }
    }
  }

  float* ob = out + (size_t)m0 * HIDDEN + n0;
#pragma unroll
  for (int n = 0; n < 4; ++n) {
    const float bias = bo[n0 + n * 16 + col];
#pragma unroll
    for (int r = 0; r < 8; ++r) {
      const int row = (lane < 16) ? r : (r + 8);
      ob[(size_t)row * HIDDEN + n * 16 + col] = acc[n][r] + bias;
    }
  }
}

// ---------------------------------------------------------------------
extern "C" void kernel_launch(void* const* d_in, const int* in_sizes, int n_in,
                              void* d_out, int out_size, void* d_ws,
                              size_t ws_size, hipStream_t stream) {
  (void)in_sizes; (void)n_in; (void)out_size; (void)ws_size;
  const float* query = (const float*)d_in[0];
  const float* key_  = (const float*)d_in[1];
  const float* value = (const float*)d_in[2];
  const int*   mask  = (const int*)d_in[3];
  const float* Wq = (const float*)d_in[4];
  const float* bq = (const float*)d_in[5];
  const float* Wk = (const float*)d_in[6];
  const float* bk = (const float*)d_in[7];
  const float* Wv = (const float*)d_in[8];
  const float* bv = (const float*)d_in[9];
  const float* Wo = (const float*)d_in[10];
  const float* bo = (const float*)d_in[11];
  float* out = (float*)d_out;

  const size_t elems = (size_t)BATCH * HEADS * SEQ * HDIM;  // 4 Mi halves
  _Float16* qf = (_Float16*)d_ws;      // [B,H,S,D] f16
  _Float16* kf = qf + elems;
  _Float16* vf = kf + elems;
  _Float16* af = vf + elems;           // [B,S,H*D] f16
  _Float16* wf = af + elems;           // Wo in f16, 1 Mi halves

  dim3 blk(256);
  head_proj_kernel<<<512, blk, 0, stream>>>(query, Wq, bq, qf);
  head_proj_kernel<<<512, blk, 0, stream>>>(key_,  Wk, bk, kf);
  head_proj_kernel<<<512, blk, 0, stream>>>(value, Wv, bv, vf);
  wcvt_kernel<<<HIDDEN * HIDDEN / 1024, blk, 0, stream>>>(Wo, wf);
  flash_attn_kernel<<<512, blk, 0, stream>>>(qf, kf, vf, mask, af);
  out_proj_kernel<<<512, blk, 0, stream>>>(af, wf, bo, out);
}
